// GraphStepLayer_62362925138389
// MI455X (gfx1250) — matrix operand order repaired and checked by reading, not compiled, not run on hardware
//
#include <hip/hip_runtime.h>
#include <hip/hip_bf16.h>
#include <math.h>

// Problem dims (fixed by reference)
#define Bq   8
#define Nq   512
#define Fq   16
#define Eq   32
#define CHSq 33
#define HATT 64
#define Mq   64
#define OUTq 64

typedef __attribute__((ext_vector_type(16))) _Float16 v16h;
typedef __attribute__((ext_vector_type(8)))  _Float16 v8h;
typedef __attribute__((ext_vector_type(8)))  float    v8f;

// ---------------------------------------------------------------------------
// Kernel 1: node encoder + attention projections.
//   enc   (B,N,E)  f32  : relu(x @ W_enc + b_enc)
//   encT  (B,E,N)  f16  : transposed f16 copy (WMMA B-operand needs K=node
//                         contiguous per output channel)
//   U,V   (B,N,64) f32  : feat_att @ W_att1[:33] (+b_att1) and @ W_att1[33:]
// One thread per node; weights are uniform -> scalar-cached loads.
// ---------------------------------------------------------------------------
__global__ void k_encode(const float* __restrict__ x,
                         const float* __restrict__ W_enc,
                         const float* __restrict__ b_enc,
                         const float* __restrict__ W_att1,
                         const float* __restrict__ b_att1,
                         float* __restrict__ enc,
                         _Float16* __restrict__ encT,
                         float* __restrict__ U,
                         float* __restrict__ V) {
    int idx = blockIdx.x * blockDim.x + threadIdx.x;
    if (idx >= Bq * Nq) return;
    int b = idx / Nq, n = idx % Nq;

    float xv[Fq];
#pragma unroll
    for (int f = 0; f < Fq; ++f) xv[f] = x[idx * Fq + f];

    float feat[CHSq];
#pragma unroll
    for (int e = 0; e < Eq; ++e) {
        float s = b_enc[e];
#pragma unroll
        for (int f = 0; f < Fq; ++f) s = fmaf(xv[f], W_enc[f * Eq + e], s);
        s = fmaxf(s, 0.0f);
        feat[e] = s;
        enc[(size_t)idx * Eq + e] = s;
        encT[((size_t)b * Eq + e) * Nq + n] = (_Float16)s;
    }
    // positional channel: arange(N)*(2/N)-1
    feat[Eq] = (float)n * (2.0f / (float)Nq) - 1.0f;

    for (int h = 0; h < HATT; ++h) {
        float su = b_att1[h];   // fold first-layer bias into U
        float sv = 0.0f;
#pragma unroll
        for (int c = 0; c < CHSq; ++c) {
            su = fmaf(feat[c], W_att1[c * HATT + h], su);
            sv = fmaf(feat[c], W_att1[(CHSq + c) * HATT + h], sv);
        }
        U[(size_t)idx * HATT + h] = su;
        V[(size_t)idx * HATT + h] = sv;
    }
}

// ---------------------------------------------------------------------------
// Kernel 2: attention scores. One block per (b,i); U row staged in LDS,
// threads stride over j.  atts[b,i,j] = b2 + sum_h relu(U+V) * w2  (f16 out).
// V (1 MB) stays L2-resident and is re-read N times -> L2 bandwidth bound,
// trivially within budget.
// ---------------------------------------------------------------------------
__global__ void k_atts(const float* __restrict__ U,
                       const float* __restrict__ V,
                       const float* __restrict__ w2,
                       const float* __restrict__ b2,
                       _Float16* __restrict__ atts) {
    int bi = blockIdx.x;                 // b*N + i
    __shared__ float sU[HATT];
    __shared__ float sw[HATT];
    if (threadIdx.x < HATT) {
        sU[threadIdx.x] = U[(size_t)bi * HATT + threadIdx.x];
        sw[threadIdx.x] = w2[threadIdx.x];
    }
    __syncthreads();
    int   b    = bi / Nq;
    float bias = b2[0];
    for (int j = threadIdx.x; j < Nq; j += blockDim.x) {
        const float* vr = V + ((size_t)b * Nq + j) * HATT;
        float s = bias;
#pragma unroll
        for (int h = 0; h < HATT; ++h)
            s = fmaf(fmaxf(sU[h] + vr[h], 0.0f), sw[h], s);
        atts[(size_t)bi * Nq + j] = (_Float16)s;
    }
}

// ---------------------------------------------------------------------------
// Kernel 3: agg = atts @ node_encs via v_wmma_f32_16x16x32_f16.
// Per wave: one 16-row tile of one batch, all 32 output channels
// (two 16-col accumulators), K loop over N=512 in steps of 32.
// A layout: lane r=lane%16 -> row i0+r; halves 0..7 = K G*8..+7,
//           halves 8..15 = K 16+G*8..+7  (two 16B loads / step).
// B layout: lane r -> col; halves 0..7 = K G*8..+7, halves 8..15 =
//           K 16+G*8..+7 from encT row (two 16B loads / step / col-tile).
// ---------------------------------------------------------------------------
__global__ void k_agg_wmma(const _Float16* __restrict__ atts,
                           const _Float16* __restrict__ encT,
                           float* __restrict__ agg) {
    int wave = threadIdx.x >> 5;
    int lane = threadIdx.x & 31;
    int tile = blockIdx.x * (blockDim.x >> 5) + wave;   // 0 .. B*N/16-1
    int b  = tile / (Nq / 16);
    int i0 = (tile % (Nq / 16)) * 16;
    int r  = lane & 15;
    int G  = lane >> 4;

    const _Float16* arow  = atts + ((size_t)(b * Nq + i0 + r)) * Nq;
    const _Float16* b0row = encT + ((size_t)(b * Eq + r)) * Nq;        // cols 0..15
    const _Float16* b1row = encT + ((size_t)(b * Eq + 16 + r)) * Nq;   // cols 16..31

    v8f c0 = {};
    v8f c1 = {};
    for (int k = 0; k < Nq; k += 32) {
        v8h alo = *(const v8h*)(arow + k + G * 8);
        v8h ahi = *(const v8h*)(arow + k + 16 + G * 8);
        v16h a;
#pragma unroll
        for (int t = 0; t < 8; ++t) { a[t] = alo[t]; a[t + 8] = ahi[t]; }

        v8h p0 = *(const v8h*)(b0row + k + G * 8);
        v8h p1 = *(const v8h*)(b0row + k + 16 + G * 8);
        v16h bm0;
#pragma unroll
        for (int t = 0; t < 8; ++t) { bm0[t] = p0[t]; bm0[t + 8] = p1[t]; }

        v8h q0 = *(const v8h*)(b1row + k + G * 8);
        v8h q1 = *(const v8h*)(b1row + k + 16 + G * 8);
        v16h bm1;
#pragma unroll
        for (int t = 0; t < 8; ++t) { bm1[t] = q0[t]; bm1[t + 8] = q1[t]; }

        c0 = __builtin_amdgcn_wmma_f32_16x16x32_f16(false, a, false, bm0,
                                                    (short)0, c0, false, false);
        c1 = __builtin_amdgcn_wmma_f32_16x16x32_f16(false, a, false, bm1,
                                                    (short)0, c1, false, false);
    }
    // C/D layout: VGPR v, lane -> row = i0 + v + 8*G, col = r / 16+r
#pragma unroll
    for (int v = 0; v < 8; ++v) {
        int row = i0 + v + 8 * G;
        agg[((size_t)(b * Nq + row)) * Eq + r]      = c0[v];
        agg[((size_t)(b * Nq + row)) * Eq + 16 + r] = c1[v];
    }
}

// ---------------------------------------------------------------------------
// Kernel 4: messages + state update. One block (64 threads) per node.
//   messages = relu(agg @ W_msg + b_msg)            (E x M)
//   out      = tanh([enc, messages] @ W_upd + b_upd)  ((E+M) x OUT)
// ---------------------------------------------------------------------------
__global__ void k_update(const float* __restrict__ enc,
                         const float* __restrict__ agg,
                         const float* __restrict__ W_msg,
                         const float* __restrict__ b_msg,
                         const float* __restrict__ W_upd,
                         const float* __restrict__ b_upd,
                         float* __restrict__ out) {
    int node = blockIdx.x;      // b*N + n
    int t    = threadIdx.x;     // 0..63
    __shared__ float sagg[Eq];
    __shared__ float senc[Eq];
    __shared__ float smsg[Mq];
    if (t < Eq) {
        sagg[t] = agg[(size_t)node * Eq + t];
        senc[t] = enc[(size_t)node * Eq + t];
    }
    __syncthreads();
    float m = b_msg[t];
#pragma unroll
    for (int e = 0; e < Eq; ++e) m = fmaf(sagg[e], W_msg[e * Mq + t], m);
    smsg[t] = fmaxf(m, 0.0f);
    __syncthreads();
    float o = b_upd[t];
#pragma unroll
    for (int c = 0; c < Eq; ++c) o = fmaf(senc[c], W_upd[c * OUTq + t], o);
#pragma unroll
    for (int mm = 0; mm < Mq; ++mm)
        o = fmaf(smsg[mm], W_upd[(Eq + mm) * OUTq + t], o);
    out[(size_t)node * OUTq + t] = tanhf(o);
}

// ---------------------------------------------------------------------------
// Workspace layout (all regions 256B-aligned, total ~7.25 MB):
//   enc_f32  : B*N*E f32   =  512 KB @ 0
//   encT_f16 : B*E*N f16   =  256 KB @ 524288
//   U        : B*N*64 f32  = 1024 KB @ 786432
//   V        : B*N*64 f32  = 1024 KB @ 1835008
//   atts_f16 : B*N*N f16   = 4096 KB @ 2883584
//   agg      : B*N*E f32   =  512 KB @ 7077888
// ---------------------------------------------------------------------------
extern "C" void kernel_launch(void* const* d_in, const int* in_sizes, int n_in,
                              void* d_out, int out_size, void* d_ws, size_t ws_size,
                              hipStream_t stream) {
    const float* x      = (const float*)d_in[0];
    const float* W_enc  = (const float*)d_in[1];
    const float* b_enc  = (const float*)d_in[2];
    const float* W_att1 = (const float*)d_in[3];
    const float* b_att1 = (const float*)d_in[4];
    const float* W_att2 = (const float*)d_in[5];
    const float* b_att2 = (const float*)d_in[6];
    const float* W_msg  = (const float*)d_in[7];
    const float* b_msg  = (const float*)d_in[8];
    const float* W_upd  = (const float*)d_in[9];
    const float* b_upd  = (const float*)d_in[10];
    float* out = (float*)d_out;

    char* ws = (char*)d_ws;
    float*    enc  = (float*)(ws + 0);
    _Float16* encT = (_Float16*)(ws + 524288);
    float*    U    = (float*)(ws + 786432);
    float*    V    = (float*)(ws + 1835008);
    _Float16* atts = (_Float16*)(ws + 2883584);
    float*    agg  = (float*)(ws + 7077888);

    k_encode<<<(Bq * Nq + 255) / 256, 256, 0, stream>>>(
        x, W_enc, b_enc, W_att1, b_att1, enc, encT, U, V);

    k_atts<<<Bq * Nq, 256, 0, stream>>>(U, V, W_att2, b_att2, atts);

    // B*N/16 = 256 wave-tiles, 4 waves (128 threads) per block -> 64 blocks
    k_agg_wmma<<<(Bq * Nq / 16) / 4, 128, 0, stream>>>(atts, encT, agg);

    k_update<<<Bq * Nq, 64, 0, stream>>>(enc, agg, W_msg, b_msg, W_upd, b_upd, out);
}
